// Attention_61795989455034
// MI455X (gfx1250) — matrix operand order repaired and checked by reading, not compiled, not run on hardware
//
#include <hip/hip_runtime.h>

typedef __attribute__((ext_vector_type(16))) _Float16 v16h;
typedef __attribute__((ext_vector_type(8)))  float    v8f;

#define BZ 64
#define SRC_LEN 2048
#define DIM 512

#define OTILES   (DIM / 16)   // 32 o-tiles
#define KCHUNKS  (DIM / 32)   // 16 k-chunks
// A-fragment store: [otile][kchunk][lane][16 halves]  (32B per lane per frag)
#define WFRAG_ELEMS (OTILES * KCHUNKS * 32 * 16)       // 262144 halves = 512KB

// ---------------------------------------------------------------------------
// Kernel 0: split W_conv (fp32) into f16 hi/lo, stored in WMMA A-fragment
// order so the GEMM consumes each fragment as an aligned v16h.
// A order used here (16x32, contiguous k, matches B): lane(lg,lr): h[0..15] -> K = 16*lg + 0..15
// ---------------------------------------------------------------------------
__global__ __launch_bounds__(256) void wsplit_kernel(
    const float* __restrict__ Wc, _Float16* __restrict__ whi,
    _Float16* __restrict__ wlo)
{
    const int idx = blockIdx.x * 256 + threadIdx.x;   // 0 .. WFRAG_ELEMS-1
    const int j     = idx & 15;
    const int lane  = (idx >> 4) & 31;
    const int kc    = (idx >> 9) & (KCHUNKS - 1);
    const int otile = idx >> 13;
    const int lr = lane & 15, lg = lane >> 4;
    const int o  = otile * 16 + lr;
    const int d  = kc * 32 + lg * 16 + j;
    const float    x  = Wc[(size_t)o * DIM + d];
    const _Float16 hi = (_Float16)x;
    whi[idx] = hi;
    wlo[idx] = (_Float16)(x - (float)hi);
}

// ---------------------------------------------------------------------------
// Kernel 1: tgt_p[b,e] = sum_d W_lin[e,d] * tgt[b,d]        (64 x 512, tiny)
// ---------------------------------------------------------------------------
__global__ __launch_bounds__(256) void tgtp_kernel(
    const float* __restrict__ tgt, const float* __restrict__ Wlin,
    float* __restrict__ tgt_p)
{
    __shared__ float t[DIM];
    const int b = blockIdx.x, tid = threadIdx.x;
    for (int i = tid; i < DIM; i += 256) t[i] = tgt[b * DIM + i];
    __syncthreads();
    for (int e = tid; e < DIM; e += 256) {
        const float* wr = Wlin + (size_t)e * DIM;
        float acc = 0.f;
        #pragma unroll 4
        for (int d = 0; d < DIM; d += 4) {
            acc += wr[d + 0] * t[d + 0];
            acc += wr[d + 1] * t[d + 1];
            acc += wr[d + 2] * t[d + 2];
            acc += wr[d + 3] * t[d + 3];
        }
        tgt_p[b * DIM + e] = acc;
    }
}

// ---------------------------------------------------------------------------
// Kernel 2: align = tgt_p . src, mask (incl. prev_idxs), softmax -> logits,
//           and mask_ output (as float 0/1). One workgroup per batch.
// ---------------------------------------------------------------------------
__global__ __launch_bounds__(256) void softmax_kernel(
    const float* __restrict__ src, const unsigned char* __restrict__ mask,
    const int* __restrict__ prev, const float* __restrict__ tgt_p,
    float* __restrict__ logits, float* __restrict__ mask_out)
{
    __shared__ float tp[DIM];
    __shared__ float red[256];
    __shared__ float alog[SRC_LEN];
    const int b = blockIdx.x, tid = threadIdx.x;
    const int pidx = prev[b];
    for (int i = tid; i < DIM; i += 256) tp[i] = tgt_p[b * DIM + i];
    __syncthreads();

    const float* srcb = src + (size_t)b * SRC_LEN * DIM;
    float lmax = -INFINITY;
    for (int s = tid; s < SRC_LEN; s += 256) {
        const float* row = srcb + (size_t)s * DIM;
        float acc = 0.f;
        #pragma unroll 4
        for (int d = 0; d < DIM; d += 4) {
            acc += row[d + 0] * tp[d + 0];
            acc += row[d + 1] * tp[d + 1];
            acc += row[d + 2] * tp[d + 2];
            acc += row[d + 3] * tp[d + 3];
        }
        const bool m = (mask[b * SRC_LEN + s] != 0) || (s == pidx);
        const float a = m ? -INFINITY : acc;
        alog[s] = a;
        lmax = fmaxf(lmax, a);
        mask_out[b * SRC_LEN + s] = m ? 1.0f : 0.0f;
    }
    red[tid] = lmax;
    __syncthreads();
    for (int off = 128; off > 0; off >>= 1) {
        if (tid < off) red[tid] = fmaxf(red[tid], red[tid + off]);
        __syncthreads();
    }
    const float mx = red[0];
    __syncthreads();

    float lsum = 0.f;
    for (int s = tid; s < SRC_LEN; s += 256) {
        const float e = __expf(alog[s] - mx);
        alog[s] = e;
        lsum += e;
    }
    red[tid] = lsum;
    __syncthreads();
    for (int off = 128; off > 0; off >>= 1) {
        if (tid < off) red[tid] += red[tid + off];
        __syncthreads();
    }
    const float inv = 1.0f / red[0];
    for (int s = tid; s < SRC_LEN; s += 256)
        logits[b * SRC_LEN + s] = alog[s] * inv;
}

// ---------------------------------------------------------------------------
// Kernel 3: attn_h[b,o,s] = sum_d Wc[o,d]*src[b,s,d] + bc[o]
// Workgroup tile: 128(o) x 128(s); 8 waves as 2(o-half) x 4(s-quarter);
// each wave computes 4(o) x 2(s) WMMA tiles (24 WMMAs / k-step).
// A (pre-split hi/lo f16 fragments) is staged global->LDS, double buffered:
// loads for k-chunk kc+1 are issued before the WMMAs of kc, drained into
// LDS at the top of the next iteration -> no exposed global-load waits.
// fp32 emulated with 3x v_wmma_f32_16x16x32_f16 (hi*hi + hi*lo + lo*hi).
// ---------------------------------------------------------------------------
__global__ __launch_bounds__(256) void conv_gemm_kernel(
    const float* __restrict__ src, const _Float16* __restrict__ whi,
    const _Float16* __restrict__ wlo, const float* __restrict__ bc,
    float* __restrict__ out)
{
    // LDS: 2 buffers x (hi 4096 halves + lo 4096 halves) = 32 KB
    __shared__ __align__(64) _Float16 abuf[2][8192];

    const int b    = blockIdx.z;
    const int ob   = blockIdx.y;        // 0..3   (o blocks of 128)
    const int sb   = blockIdx.x;        // 0..15  (s blocks of 128)
    const int tid  = threadIdx.x;
    const int wave = tid >> 5;
    const int lane = tid & 31;
    const int lg   = lane >> 4;         // lane group (0/1)
    const int lr   = lane & 15;

    const int oh  = wave & 1;           // o-half: 4 o-tiles each
    const int sq  = wave >> 1;          // s-quarter: 2 s-tiles each

    const int fr0    = oh * 4;                      // first LDS fragment index
    const int s_base = sb * 128 + sq * 32;          // wave covers 32 s values

    const float*  srcb = src + (size_t)b * SRC_LEN * DIM;
    const float4* whi4 = (const float4*)whi;        // 16B units of fragment store
    const float4* wlo4 = (const float4*)wlo;

    // copy-role indices: thread copies 32B of hi + 32B of lo per k-step
    const int cf = tid >> 5;            // fragment 0..7 (o-tile within block)
    const int ce = (tid & 31) * 2;      // float4 index within fragment (0..62)
    const int gfrag = (ob * 8 + cf) * (KCHUNKS * 64);  // float4 base of o-tile
    float4* lds4 = (float4*)&abuf[0][0];               // [buf*1024 + region]

    v8f acc[4][2];
    const v8f vzero = {0.f, 0.f, 0.f, 0.f, 0.f, 0.f, 0.f, 0.f};
    #pragma unroll
    for (int ot = 0; ot < 4; ++ot)
        #pragma unroll
        for (int st = 0; st < 2; ++st) acc[ot][st] = vzero;

    // prologue: fetch k-chunk 0 A fragments into registers
    float4 th0 = whi4[gfrag + ce];
    float4 th1 = whi4[gfrag + ce + 1];
    float4 tl0 = wlo4[gfrag + ce];
    float4 tl1 = wlo4[gfrag + ce + 1];

    for (int kc = 0; kc < KCHUNKS; ++kc) {
        const int cb = kc & 1;
        __syncthreads();   // previous reads of buffer cb are complete
        // drain the in-flight A fragments for this k-chunk into LDS
        lds4[cb * 1024 +       cf * 64 + ce]     = th0;
        lds4[cb * 1024 +       cf * 64 + ce + 1] = th1;
        lds4[cb * 1024 + 512 + cf * 64 + ce]     = tl0;
        lds4[cb * 1024 + 512 + cf * 64 + ce + 1] = tl1;
        // issue global loads for the next k-chunk (latency hidden by WMMAs)
        if (kc + 1 < KCHUNKS) {
            const int gk = gfrag + (kc + 1) * 64 + ce;
            th0 = whi4[gk];
            th1 = whi4[gk + 1];
            tl0 = wlo4[gk];
            tl1 = wlo4[gk + 1];
        }
        __syncthreads();   // LDS buffer cb is ready

        // ---- B fragments: src rows, contiguous-k B order (same as A) ----
        // lane (lg,lr): column N = lr, halves j=0..15 hold K = 16*lg + j
        const int k0 = kc * 32;
        v16h bhi[2], blo[2];
        #pragma unroll
        for (int st = 0; st < 2; ++st) {
            const float* p = srcb + (size_t)(s_base + st * 16 + lr) * DIM + k0 + lg * 16;
            if (k0 + 32 < DIM) __builtin_prefetch(p + 32, 0, 0);  // global_prefetch_b8
            #pragma unroll
            for (int j = 0; j < 16; ++j) {
                const float    x  = p[j];
                const _Float16 hi = (_Float16)x;
                bhi[st][j] = hi;
                blo[st][j] = (_Float16)(x - (float)hi);
            }
        }

        // ---- A from LDS (one fragment at a time), 3-term split WMMA ----
        const v16h* Av = (const v16h*)&abuf[cb][0];   // 512 v16h per buffer
        #pragma unroll
        for (int ot = 0; ot < 4; ++ot) {
            const v16h ahi = Av[(fr0 + ot) * 32 + lane];         // hi region
            const v16h alo = Av[256 + (fr0 + ot) * 32 + lane];   // lo region
            #pragma unroll
            for (int st = 0; st < 2; ++st) {
                acc[ot][st] = __builtin_amdgcn_wmma_f32_16x16x32_f16(
                    false, ahi, false, bhi[st], (short)0, acc[ot][st], false, false);
                acc[ot][st] = __builtin_amdgcn_wmma_f32_16x16x32_f16(
                    false, ahi, false, blo[st], (short)0, acc[ot][st], false, false);
                acc[ot][st] = __builtin_amdgcn_wmma_f32_16x16x32_f16(
                    false, alo, false, bhi[st], (short)0, acc[ot][st], false, false);
            }
        }
    }

    // ---- store: C/D layout -> VGPR r: row M = r + 8*lg, col N = lr ----
    const int otile0 = ob * 8 + oh * 4;
    #pragma unroll
    for (int ot = 0; ot < 4; ++ot) {
        #pragma unroll
        for (int r = 0; r < 8; ++r) {
            const int   o    = (otile0 + ot) * 16 + r + 8 * lg;
            const float bias = bc[o];
            float* orow = out + ((size_t)b * DIM + o) * SRC_LEN;
            #pragma unroll
            for (int st = 0; st < 2; ++st) {
                const int s = s_base + st * 16 + lr;
                orow[s] = acc[ot][st][r] + bias;
            }
        }
    }
}

// ---------------------------------------------------------------------------
extern "C" void kernel_launch(void* const* d_in, const int* in_sizes, int n_in,
                              void* d_out, int out_size, void* d_ws, size_t ws_size,
                              hipStream_t stream)
{
    const float*         src   = (const float*)d_in[0];
    const float*         tgt   = (const float*)d_in[1];
    const unsigned char* mask  = (const unsigned char*)d_in[2];
    const int*           prev  = (const int*)d_in[3];
    const float*         Wlin  = (const float*)d_in[4];
    // d_in[5] = W_out: feeds only dead code (attn_h is overwritten) -> unused
    const float*         Wconv = (const float*)d_in[6];
    const float*         bconv = (const float*)d_in[7];

    float* out      = (float*)d_out;
    float* attn_h   = out;                                       // (64,512,2048)
    float* logits   = out + (size_t)BZ * DIM * SRC_LEN;          // (64,1,2048)
    float* mask_out = logits + (size_t)BZ * SRC_LEN;             // (64,1,2048)

    // workspace layout: tgt_p (128KB) | w_hi (512KB) | w_lo (512KB)
    float*    tgt_p = (float*)d_ws;
    _Float16* w_hi  = (_Float16*)((char*)d_ws + (size_t)BZ * DIM * sizeof(float));
    _Float16* w_lo  = w_hi + WFRAG_ELEMS;

    wsplit_kernel<<<WFRAG_ELEMS / 256, 256, 0, stream>>>(Wconv, w_hi, w_lo);
    tgtp_kernel<<<BZ, 256, 0, stream>>>(tgt, Wlin, tgt_p);
    softmax_kernel<<<BZ, 256, 0, stream>>>(src, mask, prev, tgt_p, logits, mask_out);

    dim3 grid(SRC_LEN / 128, DIM / 128, BZ);   // (16, 4, 64)
    conv_gemm_kernel<<<grid, 256, 0, stream>>>(src, w_hi, w_lo, bconv, attn_h);
}
